// GCNNet_63376537419956
// MI455X (gfx1250) — matrix operand order repaired and checked by reading, not compiled, not run on hardware
//
#include <hip/hip_runtime.h>
#include <math.h>

#define NNODES 4096
#define FDIM   64
#define NGRAPH 16
#define NHEADS 8
#define CDIV(a,b) (((a)+(b)-1)/(b))

typedef __attribute__((ext_vector_type(16))) _Float16 v16h;
typedef __attribute__((ext_vector_type(8)))  float    v8f;

// ---------- helpers ----------
__device__ __forceinline__ unsigned fkey(float f) {
  unsigned u = __float_as_uint(f);
  return (u & 0x80000000u) ? ~u : (u | 0x80000000u);   // order-preserving float->uint
}
__device__ __forceinline__ float funkey(unsigned k) {
  unsigned u = (k & 0x80000000u) ? (k & 0x7fffffffu) : ~k;
  return __uint_as_float(u);
}
__device__ __forceinline__ float sigmf(float x) { return 1.0f / (1.0f + __expf(-x)); }

__device__ __forceinline__ void edge_nodes(const int* __restrict__ ei, int E, int e,
                                           int& s, int& d) {
  if (e < E) { s = ei[e]; d = ei[E + e]; } else { s = d = e - E; }  // appended self-loops
}

// ---------- weight pre-transpose: W[KxN] f32 -> Wt[NxK] f16 ----------
__global__ void k_convw(const float* __restrict__ W, _Float16* __restrict__ Wt,
                        int K, int N) {
  int i = blockIdx.x * blockDim.x + threadIdx.x;
  if (i >= K * N) return;
  int k = i / N, n = i % N;
  Wt[(size_t)n * K + k] = (_Float16)W[i];
}

// ---------- WMMA GEMM: C = act(A[MxK] * W[KxN] + bias) ----------
// Wt is the f16, [N][K]-transposed weight. One wave computes a 16x64 strip
// (4 WMMA accumulators sharing one A fragment). act: 0=none 1=relu 2=elu
__global__ __launch_bounds__(32)
void k_gemm_wmma(const float* __restrict__ A, const _Float16* __restrict__ Wt,
                 const float* __restrict__ bias, float* __restrict__ C,
                 int M, int K, int N, int act) {
  const int lane = threadIdx.x;
  const int tm = blockIdx.x, tn = blockIdx.y;
  const bool hi = lane >= 16;
  const int row  = tm * 16 + (lane & 15);
  const int col0 = tn * 64 + (lane & 15);
  const int aklo = hi ? 16 : 0;     // A K-half offset within 32
  const int bkb  = hi ? 16 : 0;     // B K-half offset

  const _Float16* Wp[4];
#pragma unroll
  for (int t = 0; t < 4; ++t) {
    int c = col0 + 16 * t;
    int cc = (c < N) ? c : (N - 1);          // clamp: OOB columns computed, never stored
    Wp[t] = Wt + (size_t)cc * K + bkb;
  }

  v8f c0 = {}, c1 = {}, c2 = {}, c3 = {};
  const float* Ap = A + (size_t)row * K;
  for (int k0 = 0; k0 < K; k0 += 32) {
    if (k0 + 32 < K) __builtin_prefetch(Ap + k0 + 32, 0, 1);   // global_prefetch_b8
    v16h a;
#pragma unroll
    for (int i = 0; i < 8; ++i) {
      a[i]     = (_Float16)Ap[k0 + aklo + i];
      a[i + 8] = (_Float16)Ap[k0 + aklo + 8 + i];
    }
    v16h b0 = *(const v16h*)(Wp[0] + k0);
    v16h b1 = *(const v16h*)(Wp[1] + k0);
    v16h b2 = *(const v16h*)(Wp[2] + k0);
    v16h b3 = *(const v16h*)(Wp[3] + k0);
    c0 = __builtin_amdgcn_wmma_f32_16x16x32_f16(false, a, false, b0, (short)0, c0, false, false);
    c1 = __builtin_amdgcn_wmma_f32_16x16x32_f16(false, a, false, b1, (short)0, c1, false, false);
    c2 = __builtin_amdgcn_wmma_f32_16x16x32_f16(false, a, false, b2, (short)0, c2, false, false);
    c3 = __builtin_amdgcn_wmma_f32_16x16x32_f16(false, a, false, b3, (short)0, c3, false, false);
  }

  const int rbase = tm * 16 + (hi ? 8 : 0);
  v8f accs[4] = {c0, c1, c2, c3};
#pragma unroll
  for (int t = 0; t < 4; ++t) {
    int c = col0 + 16 * t;
    if (c < N) {
      const float bv = bias ? bias[c] : 0.0f;
#pragma unroll
      for (int j = 0; j < 8; ++j) {
        float v = accs[t][j] + bv;
        if (act == 1)      v = fmaxf(v, 0.0f);
        else if (act == 2) v = (v > 0.0f) ? v : (__expf(v) - 1.0f);
        C[(size_t)(rbase + j) * N + c] = v;
      }
    }
  }
}

// ---------- elementwise ----------
__global__ void k_zero(float* p, int n) {
  int i = blockIdx.x * blockDim.x + threadIdx.x;
  if (i < n) p[i] = 0.0f;     // bit pattern 0 also valid for uint keys
}
__global__ void k_bias_act(const float* __restrict__ acc, const float* __restrict__ bias,
                           float* __restrict__ out, int total, int N, int act) {
  int i = blockIdx.x * blockDim.x + threadIdx.x;
  if (i >= total) return;
  float v = acc[i] + bias[i % N];
  if (act == 1)      v = fmaxf(v, 0.0f);
  else if (act == 2) v = (v > 0.0f) ? v : (__expf(v) - 1.0f);
  out[i] = v;
}

// ---------- GAT attention ----------
__global__ void k_att_coef(const float* __restrict__ h, const float* __restrict__ as,
                           const float* __restrict__ ad, float* __restrict__ al,
                           float* __restrict__ ar, int Nn, int Hh, int D) {
  int i = blockIdx.x * blockDim.x + threadIdx.x;
  if (i >= Nn * Hh) return;
  int n = i / Hh, hh = i % Hh;
  const float* hp = h + ((size_t)n * Hh + hh) * D;
  float l = 0.f, r = 0.f;
  for (int d = 0; d < D; ++d) { l += hp[d] * as[hh * D + d]; r += hp[d] * ad[hh * D + d]; }
  al[i] = l; ar[i] = r;
}
__device__ __forceinline__ float att_logit(const float* al, const float* ar,
                                           int s, int d, int hh, int Hh) {
  float l = al[s * Hh + hh] + ar[d * Hh + hh];
  return l > 0.f ? l : 0.2f * l;   // leaky_relu(0.2)
}
__global__ void k_att_max(const int* ei, int E, int Nn, const float* al, const float* ar,
                          unsigned* mkey, int Hh) {
  int i = blockIdx.x * blockDim.x + threadIdx.x;
  int tot = (E + Nn) * Hh;
  if (i >= tot) return;
  int e = i / Hh, hh = i % Hh, s, d;
  edge_nodes(ei, E, e, s, d);
  atomicMax(&mkey[d * Hh + hh], fkey(att_logit(al, ar, s, d, hh, Hh)));
}
__global__ void k_att_sum(const int* ei, int E, int Nn, const float* al, const float* ar,
                          const unsigned* mkey, float* sden, int Hh) {
  int i = blockIdx.x * blockDim.x + threadIdx.x;
  int tot = (E + Nn) * Hh;
  if (i >= tot) return;
  int e = i / Hh, hh = i % Hh, s, d;
  edge_nodes(ei, E, e, s, d);
  float l = att_logit(al, ar, s, d, hh, Hh);
  atomicAdd(&sden[d * Hh + hh], __expf(l - funkey(mkey[d * Hh + hh])));
}
__global__ void k_att_agg(const int* ei, int E, int Nn, const float* al, const float* ar,
                          const unsigned* mkey, const float* sden,
                          const float* __restrict__ hfeat, float* __restrict__ acc,
                          int Hh, int D) {
  int i = blockIdx.x * blockDim.x + threadIdx.x;
  int tot = (E + Nn) * Hh;
  if (i >= tot) return;
  int e = i / Hh, hh = i % Hh, s, d;
  edge_nodes(ei, E, e, s, d);
  float l = att_logit(al, ar, s, d, hh, Hh);
  float alpha = __expf(l - funkey(mkey[d * Hh + hh])) / sden[d * Hh + hh];
  const float* hs = hfeat + ((size_t)s * Hh + hh) * D;
  float* ad2 = acc + ((size_t)d * Hh + hh) * D;
  for (int k = 0; k < D; ++k) atomicAdd(&ad2[k], alpha * hs[k]);
}

// ---------- GCN ----------
__global__ void k_deg(const int* ei, int E, int Nn, float* deg) {
  int e = blockIdx.x * blockDim.x + threadIdx.x;
  if (e >= E + Nn) return;
  int s, d; edge_nodes(ei, E, e, s, d);
  atomicAdd(&deg[d], 1.0f);
}
__global__ void k_gcn_agg(const int* ei, int E, int Nn, const float* __restrict__ deg,
                          const float* __restrict__ h, float* __restrict__ acc, int K) {
  int e = blockIdx.x * blockDim.x + threadIdx.x;
  if (e >= E + Nn) return;
  int s, d; edge_nodes(ei, E, e, s, d);
  float norm = rsqrtf(deg[s]) * rsqrtf(deg[d]);
  const float* hs = h + (size_t)s * K;
  float* ad = acc + (size_t)d * K;
  for (int k = 0; k < K; ++k) atomicAdd(&ad[k], norm * hs[k]);
}

// ---------- gate / GRU / pool ----------
__global__ void k_gate(const float* a_, const float* b_, float* hg, int n) {
  int i = blockIdx.x * blockDim.x + threadIdx.x;
  if (i >= n) return;
  float z = sigmf(a_[i] + b_[i]);
  hg[i] = z * b_[i] + (1.0f - z) * a_[i];
}
__global__ void k_gru(const float* gi, const float* gh, const float* hg, float* cc, int Nn, int F) {
  int i = blockIdx.x * blockDim.x + threadIdx.x;
  if (i >= Nn * F) return;
  int n = i / F, f = i % F;
  const float* gin = gi + (size_t)n * 3 * F;
  const float* ghn = gh + (size_t)n * 3 * F;
  float r  = sigmf(gin[f] + ghn[f]);
  float zz = sigmf(gin[F + f] + ghn[F + f]);
  float nn = tanhf(gin[2 * F + f] + r * ghn[2 * F + f]);
  cc[i] = (1.0f - zz) * nn + zz * hg[i];
}
__global__ void k_pool_max(const float* cc, float* g, int per, int F, int nb) {
  int i = blockIdx.x * blockDim.x + threadIdx.x;
  if (i >= nb * F) return;
  int b = i / F, f = i % F;
  float m = -3.4e38f;
  for (int k = 0; k < per; ++k) m = fmaxf(m, cc[(size_t)(b * per + k) * F + f]);
  g[i] = m;
}

// ---------- host orchestration ----------
static void gemm(hipStream_t st, const float* A, const _Float16* Wt, const float* b,
                 float* C, int M, int K, int N, int act) {
  dim3 g(M / 16, CDIV(N, 64));
  k_gemm_wmma<<<g, 32, 0, st>>>(A, Wt, b, C, M, K, N, act);
}
static void zero(hipStream_t st, float* p, int n) {
  k_zero<<<CDIV(n, 256), 256, 0, st>>>(p, n);
}

struct GatW {
  const _Float16 *g1_Wt, *g2_Wt, *gfc_Wt, *gfc1_Wt, *gout_Wt;
  const float *g1_as, *g1_ad, *g1_b, *g2_as, *g2_ad, *g2_b;
  const float *gfc_b, *gfc1_b, *gout_b;
};
struct Scratch {
  float *hf, *t1, *acc, *al, *ar, *sden, *t2, *t3, *t4;
  unsigned* mkey;
};

static void attention(hipStream_t st, const int* ei, int E, const float* hfeat,
                      const float* as, const float* ad, Scratch& w, int Hh, int D,
                      float* accout) {
  const int N = NNODES;
  k_att_coef<<<CDIV(N * Hh, 256), 256, 0, st>>>(hfeat, as, ad, w.al, w.ar, N, Hh, D);
  zero(st, (float*)w.mkey, N * Hh);
  zero(st, w.sden, N * Hh);
  zero(st, accout, N * Hh * D);
  int tot = (E + N) * Hh;
  k_att_max<<<CDIV(tot, 256), 256, 0, st>>>(ei, E, N, w.al, w.ar, w.mkey, Hh);
  k_att_sum<<<CDIV(tot, 256), 256, 0, st>>>(ei, E, N, w.al, w.ar, w.mkey, w.sden, Hh);
  k_att_agg<<<CDIV(tot, 256), 256, 0, st>>>(ei, E, N, w.al, w.ar, w.mkey, w.sden,
                                            hfeat, accout, Hh, D);
}

static void gat_block(hipStream_t st, const GatW& P, Scratch& w,
                      const float* xin, const int* ei, int E, float* out) {
  const int N = NNODES;
  gemm(st, xin, P.g1_Wt, nullptr, w.hf, N, FDIM, NHEADS * FDIM, 0);      // [N,512]
  attention(st, ei, E, w.hf, P.g1_as, P.g1_ad, w, NHEADS, FDIM, w.acc);
  k_bias_act<<<CDIV(N * NHEADS * FDIM, 256), 256, 0, st>>>(w.acc, P.g1_b, w.t1,
                                                           N * NHEADS * FDIM, NHEADS * FDIM, 2);
  gemm(st, w.t1, P.g2_Wt, nullptr, w.hf, N, NHEADS * FDIM, FDIM, 0);     // [N,64]
  attention(st, ei, E, w.hf, P.g2_as, P.g2_ad, w, 1, FDIM, w.acc);
  k_bias_act<<<CDIV(N * FDIM, 256), 256, 0, st>>>(w.acc, P.g2_b, w.t2, N * FDIM, FDIM, 1);
  gemm(st, w.t2, P.gfc_Wt, P.gfc_b, w.t3, N, FDIM, FDIM, 1);
  gemm(st, w.t3, P.gfc1_Wt, P.gfc1_b, w.t4, N, FDIM, 128, 1);
  gemm(st, w.t4, P.gout_Wt, P.gout_b, out, N, 128, FDIM, 1);  // outer relu fused
}

extern "C" void kernel_launch(void* const* d_in, const int* in_sizes, int n_in,
                              void* d_out, int out_size, void* d_ws, size_t ws_size,
                              hipStream_t stream) {
  const int N = NNODES;
  const float* x  = (const float*)d_in[0];
  const int* e1 = (const int*)d_in[1]; int E1 = in_sizes[1] / 2;
  const int* e2 = (const int*)d_in[2]; int E2 = in_sizes[2] / 2;
  const int* e3 = (const int*)d_in[3]; int E3 = in_sizes[3] / 2;
  int pi = 5;
  const float* g1_W  = (const float*)d_in[pi++]; const float* g1_as = (const float*)d_in[pi++];
  const float* g1_ad = (const float*)d_in[pi++]; const float* g1_b  = (const float*)d_in[pi++];
  const float* g2_W  = (const float*)d_in[pi++]; const float* g2_as = (const float*)d_in[pi++];
  const float* g2_ad = (const float*)d_in[pi++]; const float* g2_b  = (const float*)d_in[pi++];
  const float* gfc_W = (const float*)d_in[pi++]; const float* gfc_b = (const float*)d_in[pi++];
  const float* gfc1_W= (const float*)d_in[pi++]; const float* gfc1_b= (const float*)d_in[pi++];
  const float* gout_W= (const float*)d_in[pi++]; const float* gout_b= (const float*)d_in[pi++];
  const float* c2_W  = (const float*)d_in[pi++]; const float* c2_b  = (const float*)d_in[pi++];
  const float* c3_W  = (const float*)d_in[pi++]; const float* c3_b  = (const float*)d_in[pi++];
  const float* A_W   = (const float*)d_in[pi++]; const float* A_b   = (const float*)d_in[pi++];
  const float* B_W   = (const float*)d_in[pi++]; const float* B_b   = (const float*)d_in[pi++];
  const float* Wih   = (const float*)d_in[pi++]; const float* Whh   = (const float*)d_in[pi++];
  const float* bih   = (const float*)d_in[pi++]; const float* bhh   = (const float*)d_in[pi++];
  const float* f1_W  = (const float*)d_in[pi++]; const float* f1_b  = (const float*)d_in[pi++];
  const float* f2_W  = (const float*)d_in[pi++]; const float* f2_b  = (const float*)d_in[pi++];

  float* p = (float*)d_ws;
  auto alloc = [&](size_t n) { float* r = p; p += n; return r; };
  Scratch w;
  w.hf   = alloc((size_t)N * 512);
  w.t1   = alloc((size_t)N * 512);
  w.acc  = alloc((size_t)N * 512);
  w.al   = alloc((size_t)N * NHEADS);
  w.ar   = alloc((size_t)N * NHEADS);
  w.mkey = (unsigned*)alloc((size_t)N * NHEADS);
  w.sden = alloc((size_t)N * NHEADS);
  w.t2   = alloc((size_t)N * 64);
  w.t3   = alloc((size_t)N * 64);
  w.t4   = alloc((size_t)N * 128);
  float* h1 = alloc((size_t)N * 64);
  float* h2 = alloc((size_t)N * 128);
  float* h3 = alloc((size_t)N * 256);
  float* h4 = alloc((size_t)N * 64);
  float* h5 = alloc((size_t)N * 128);
  float* h6 = alloc((size_t)N * 64);
  float* deg = alloc(N);
  float* a_ = alloc((size_t)N * 64);
  float* b_ = alloc((size_t)N * 64);
  float* hg = alloc((size_t)N * 64);
  float* gi = alloc((size_t)N * 192);
  float* gh = alloc((size_t)N * 192);
  float* cc = alloc((size_t)N * 64);
  float* gpool = alloc(NGRAPH * 64);
  float* gmid  = alloc(NGRAPH * 128);

  // f16 transposed weight staging (emitted once per launch; tiny)
  _Float16* hp16 = (_Float16*)p;
  auto convw = [&](const float* W, int K, int Nc) {
    _Float16* t = hp16; hp16 += (size_t)K * Nc;
    k_convw<<<CDIV(K * Nc, 256), 256, 0, stream>>>(W, t, K, Nc);
    return (const _Float16*)t;
  };
  const _Float16* g1_Wt   = convw(g1_W, 64, 512);
  const _Float16* g2_Wt   = convw(g2_W, 512, 64);
  const _Float16* gfc_Wt  = convw(gfc_W, 64, 64);
  const _Float16* gfc1_Wt = convw(gfc1_W, 64, 128);
  const _Float16* gout_Wt = convw(gout_W, 128, 64);
  const _Float16* c2_Wt   = convw(c2_W, 64, 128);
  const _Float16* c3_Wt   = convw(c3_W, 128, 256);
  const _Float16* A_Wt    = convw(A_W, 256, 64);
  const _Float16* B_Wt    = convw(B_W, 128, 64);
  const _Float16* Wih_t   = convw(Wih, 64, 192);
  const _Float16* Whh_t   = convw(Whh, 64, 192);
  const _Float16* f1_Wt   = convw(f1_W, 64, 128);
  const _Float16* f2_Wt   = convw(f2_W, 128, 120);

  GatW P = {g1_Wt, g2_Wt, gfc_Wt, gfc1_Wt, gout_Wt,
            g1_as, g1_ad, g1_b, g2_as, g2_ad, g2_b,
            gfc_b, gfc1_b, gout_b};

  // three GAT blocks (relu fused into final gemm of each block)
  gat_block(stream, P, w, x, e1, E1, h1);
  gat_block(stream, P, w, x, e2, E2, h4);
  gat_block(stream, P, w, x, e3, E3, h6);

  // GCN chain on e1: h2 = relu(gcn(h1,c2)), h3 = relu(gcn(h2,c3))
  zero(stream, deg, N);
  k_deg<<<CDIV(E1 + N, 256), 256, 0, stream>>>(e1, E1, N, deg);
  gemm(stream, h1, c2_Wt, nullptr, w.hf, N, 64, 128, 0);
  zero(stream, w.acc, N * 128);
  k_gcn_agg<<<CDIV(E1 + N, 256), 256, 0, stream>>>(e1, E1, N, deg, w.hf, w.acc, 128);
  k_bias_act<<<CDIV(N * 128, 256), 256, 0, stream>>>(w.acc, c2_b, h2, N * 128, 128, 1);

  gemm(stream, h2, c3_Wt, nullptr, w.hf, N, 128, 256, 0);
  zero(stream, w.acc, N * 256);
  k_gcn_agg<<<CDIV(E1 + N, 256), 256, 0, stream>>>(e1, E1, N, deg, w.hf, w.acc, 256);
  k_bias_act<<<CDIV(N * 256, 256), 256, 0, stream>>>(w.acc, c3_b, h3, N * 256, 256, 1);

  // h5 = relu(gcn(h4, e2, c2))
  zero(stream, deg, N);
  k_deg<<<CDIV(E2 + N, 256), 256, 0, stream>>>(e2, E2, N, deg);
  gemm(stream, h4, c2_Wt, nullptr, w.hf, N, 64, 128, 0);
  zero(stream, w.acc, N * 128);
  k_gcn_agg<<<CDIV(E2 + N, 256), 256, 0, stream>>>(e2, E2, N, deg, w.hf, w.acc, 128);
  k_bias_act<<<CDIV(N * 128, 256), 256, 0, stream>>>(w.acc, c2_b, h5, N * 128, 128, 1);

  // GateUpdate
  gemm(stream, h5, B_Wt, B_b, b_, N, 128, 64, 0);
  gemm(stream, h3, A_Wt, A_b, a_, N, 256, 64, 0);
  k_gate<<<CDIV(N * 64, 256), 256, 0, stream>>>(a_, b_, hg, N * 64);

  // GRUCell(h6, hg)
  gemm(stream, h6, Wih_t, bih, gi, N, 64, 192, 0);
  gemm(stream, hg, Whh_t, bhh, gh, N, 64, 192, 0);
  k_gru<<<CDIV(N * 64, 256), 256, 0, stream>>>(gi, gh, hg, cc, N, 64);

  // global_max_pool (contiguous batch blocks of N/B nodes) + MLP head
  k_pool_max<<<CDIV(NGRAPH * 64, 256), 256, 0, stream>>>(cc, gpool, N / NGRAPH, 64, NGRAPH);
  gemm(stream, gpool, f1_Wt, f1_b, gmid, NGRAPH, 64, 128, 1);
  gemm(stream, gmid, f2_Wt, f2_b, (float*)d_out, NGRAPH, 128, 120, 0);
}
